// DSMNet_34926674051231
// MI455X (gfx1250) — compile-verified, matrix operand order checked by reading, not run on hardware
//
#include <hip/hip_runtime.h>
#include <hip/hip_bf16.h>

typedef __attribute__((ext_vector_type(16))) __bf16 v16bf;
typedef __attribute__((ext_vector_type(8)))  float  v8f;
typedef __attribute__((ext_vector_type(4)))  float  f4;

#define C_   32
#define D_   48
#define H_   64
#define W_   208
#define HW_  (H_ * W_)     // 13312
#define DHW_ (D_ * HW_)    // 638976
#define TOT_ (C_ * DHW_)   // 20447232
#define EPS_ 1e-5f

// ---------------------------------------------------------------------------
// 1) L1-normalize guidance weights: 4 dirs x 5 weights per (h,w)
// ---------------------------------------------------------------------------
__global__ void knorm_kernel(const float* __restrict__ g, float* __restrict__ kn) {
  int t = blockIdx.x * blockDim.x + threadIdx.x;
  if (t >= 4 * HW_) return;
  int dir = t / HW_, pix = t % HW_;
  const float* gp = g + (dir * 5) * HW_ + pix;
  float v0 = gp[0 * HW_], v1 = gp[1 * HW_], v2 = gp[2 * HW_],
        v3 = gp[3 * HW_], v4 = gp[4 * HW_];
  float s = fabsf(v0) + fabsf(v1) + fabsf(v2) + fabsf(v3) + fabsf(v4);
  float inv = 1.f / fmaxf(s, 1e-12f);
  float* kp = kn + (dir * 5) * HW_ + pix;
  kp[0 * HW_] = v0 * inv; kp[1 * HW_] = v1 * inv; kp[2 * HW_] = v2 * inv;
  kp[3 * HW_] = v3 * inv; kp[4 * HW_] = v4 * inv;
}

// ---------------------------------------------------------------------------
// 2) Transpose x: [c][d][h][w] -> xt: [c][h][w][d]  (d fastest for the scans)
// ---------------------------------------------------------------------------
__global__ void transpose_kernel(const float* __restrict__ x, float* __restrict__ xt) {
  __shared__ float tile[48 * 33];
  int c = blockIdx.x / H_, h = blockIdx.x % H_;
  const float* xp = x + (size_t)(c * D_) * HW_ + h * W_;  // xp[d*HW_ + w]
  float* op = xt + (size_t)(c * H_ + h) * W_ * 48;        // op[w*48 + d]
  for (int w0 = 0; w0 < W_; w0 += 32) {
    for (int i = threadIdx.x; i < 48 * 32; i += blockDim.x) {
      int d = i >> 5, wl = i & 31, w = w0 + wl;
      if (w < W_) tile[d * 33 + wl] = xp[(size_t)d * HW_ + w];
    }
    __syncthreads();
    for (int i = threadIdx.x; i < 48 * 32; i += blockDim.x) {
      int d = i % 48, wl = i / 48, w = w0 + wl;
      if (w < W_) op[w * 48 + d] = tile[d * 33 + wl];
    }
    __syncthreads();
  }
}

// ---------------------------------------------------------------------------
// 3) SGA directional scans. One block per scan line, threads = disparities.
//    A[p,d] = w0*x + w1*A[p-1,d] + w2*A[p-1,d-1] + w3*A[p-1,d+1] + w4*max_d A[p-1]
//    carry init = raw x slice at p=0 (matches jax.lax.scan in reference).
// ---------------------------------------------------------------------------
__device__ __forceinline__ void sga_step(float xi, float prev, bool act, int d,
                                         float w0v, float w1v, float w2v, float w3v, float w4v,
                                         float* sA, float* red, float& outv) {
  if (act) sA[d] = prev;
  __syncthreads();
  if (threadIdx.x < 16) {
    float m = fmaxf(sA[threadIdx.x], fmaxf(sA[threadIdx.x + 16], sA[threadIdx.x + 32]));
    red[threadIdx.x] = m;
  }
  __syncthreads();
  float mx = red[0];
#pragma unroll
  for (int j = 1; j < 16; ++j) mx = fmaxf(mx, red[j]);
  float up = 0.f, dn = 0.f;
  if (act) {
    up = sA[d == 0 ? 0 : d - 1];
    dn = sA[d == 47 ? 47 : d + 1];
  }
  outv = w0v * xi + w1v * prev + w2v * up + w3v * dn + w4v * mx;
  __syncthreads();
}

__global__ void sga_scan_h(const float* __restrict__ xt, const float* __restrict__ kn,
                           float* __restrict__ sga, int dir, int rev, int combine) {
  __shared__ float sA[48];
  __shared__ float red[16];
  int c = blockIdx.x / H_, h = blockIdx.x % H_;
  int d = threadIdx.x;
  bool act = d < 48;
  const float* kb = kn + (dir * 5) * HW_ + h * W_;
  const float* xb = xt + (size_t)(c * H_ + h) * W_ * 48;
  float* ob = sga + (size_t)(c * H_ + h) * W_ * 48;
  float prev = 0.f;
  for (int p = 0; p < W_; ++p) {
    int w = rev ? (W_ - 1 - p) : p;
    float w0v = kb[w], w1v = kb[HW_ + w], w2v = kb[2 * HW_ + w],
          w3v = kb[3 * HW_ + w], w4v = kb[4 * HW_ + w];
    float xi = act ? xb[w * 48 + d] : 0.f;
    if (p == 0) prev = xi;
    float outv;
    sga_step(xi, prev, act, d, w0v, w1v, w2v, w3v, w4v, sA, red, outv);
    prev = outv;
    if (act) {
      int idx = w * 48 + d;
      float o = outv;
      if (combine) o = fmaxf(o, ob[idx]);
      ob[idx] = o;
    }
  }
}

__global__ void sga_scan_v(const float* __restrict__ xt, const float* __restrict__ kn,
                           float* __restrict__ sga, int dir, int rev, int combine) {
  __shared__ float sA[48];
  __shared__ float red[16];
  int c = blockIdx.x / W_, w = blockIdx.x % W_;
  int d = threadIdx.x;
  bool act = d < 48;
  float prev = 0.f;
  for (int p = 0; p < H_; ++p) {
    int h = rev ? (H_ - 1 - p) : p;
    const float* kp = kn + (dir * 5) * HW_ + h * W_ + w;
    float w0v = kp[0], w1v = kp[HW_], w2v = kp[2 * HW_],
          w3v = kp[3 * HW_], w4v = kp[4 * HW_];
    size_t base = (size_t)(c * H_ + h) * W_ * 48 + (size_t)w * 48;
    float xi = act ? xt[base + d] : 0.f;
    if (p == 0) prev = xi;
    float outv;
    sga_step(xi, prev, act, d, w0v, w1v, w2v, w3v, w4v, sA, red, outv);
    prev = outv;
    if (act) {
      float o = outv;
      if (combine) o = fmaxf(o, sga[base + d]);
      sga[base + d] = o;
    }
  }
}

// ---------------------------------------------------------------------------
// 4) BN1 + ReLU + cast to bf16, repack [c][h][w][d] -> NDHWC [d][h][w][c]
// ---------------------------------------------------------------------------
__global__ void bn1_relu_pack(const float* __restrict__ sga,
                              const float* __restrict__ g1, const float* __restrict__ b1,
                              const float* __restrict__ m1, const float* __restrict__ v1,
                              __bf16* __restrict__ actb) {
  __shared__ float lds[32 * 49];
  int pix = blockIdx.x;  // h*W_ + w
  for (int i = threadIdx.x; i < 32 * 48; i += blockDim.x) {
    int c = i / 48, d = i % 48;
    float s = g1[c] * rsqrtf(v1[c] + EPS_);
    float b = b1[c] - m1[c] * s;
    float val = sga[(size_t)c * DHW_ + (size_t)pix * 48 + d] * s + b;
    lds[c * 49 + d] = fmaxf(val, 0.f);
  }
  __syncthreads();
  for (int i = threadIdx.x; i < 32 * 48; i += blockDim.x) {
    int c = i & 31, d = i >> 5;
    actb[((size_t)d * HW_ + pix) * 32 + c] = (__bf16)lds[c * 49 + d];
  }
}

// ---------------------------------------------------------------------------
// 5) Pre-swizzle conv weights into WMMA B-fragment order.
//    wf[tap][nhalf][lane][16 halves], K = cin, N = cout-half.
//    Per-lane K order mirrors 16-bit fragment layout:
//      lanes 0-15: K=0..7 then 16..23 ; lanes 16-31: K=8..15 then 24..31
// ---------------------------------------------------------------------------
__global__ void wpack_kernel(const float* __restrict__ cw, __bf16* __restrict__ wf) {
  int t = blockIdx.x * blockDim.x + threadIdx.x;
  if (t >= 27 * 2 * 32) return;
  int tap = t / 64, r = t % 64, nh = r >> 5, lane = r & 31;
  int n = (lane & 15) + nh * 16;
  int gsel = lane >> 4;
  __bf16* o = wf + t * 16;
#pragma unroll
  for (int j = 0; j < 16; ++j) {
    int k = (j < 8) ? (gsel * 8 + j) : (16 + gsel * 8 + (j - 8));
    o[j] = (__bf16)cw[((size_t)n * 32 + k) * 27 + tap];
  }
}

// ---------------------------------------------------------------------------
// 6) Implicit-GEMM conv3d 3x3x3 via v_wmma_f32_16x16x32_bf16,
//    fused BN2 + residual + ReLU epilogue.
//    One wave per tile: M = 16 consecutive w, N = 32 couts (2 halves),
//    K-loop: 27 taps x 32 cin -> 54 WMMAs.
// ---------------------------------------------------------------------------
__global__ __launch_bounds__(256, 2) void conv_bn2_res_kernel(
    const __bf16* __restrict__ actb, const __bf16* __restrict__ wf,
    const float* __restrict__ xin,
    const float* __restrict__ g2, const float* __restrict__ b2,
    const float* __restrict__ m2, const float* __restrict__ v2,
    float* __restrict__ out) {
  __shared__ float lds[8 * 512];
  int wave = threadIdx.x >> 5;
  int lane = threadIdx.x & 31;
  int tile = blockIdx.x * 8 + wave;                 // 39936 tiles exactly
  int s0 = tile * 16;
  int dz = s0 / HW_;
  int rr0 = s0 - dz * HW_;
  int hy = rr0 / W_;
  int w0 = rr0 - hy * W_;                           // multiple of 16
  int m = lane & 15, gsel = lane >> 4;

  v8f acc0 = {}; v8f acc1 = {};

  for (int kd = 0; kd < 3; ++kd) {
    int dd = dz + kd - 1;
    bool dok = (unsigned)dd < (unsigned)D_;
    for (int kh = 0; kh < 3; ++kh) {
      int hh = hy + kh - 1;
      bool hok = dok && ((unsigned)hh < (unsigned)H_);
#pragma unroll
      for (int kw = 0; kw < 3; ++kw) {
        int tap = kd * 9 + kh * 3 + kw;
        const __bf16* wb = wf + (size_t)(tap * 64) * 16;
        if (tap < 26) __builtin_prefetch(wf + (size_t)((tap + 1) * 64) * 16, 0, 0);
        v16bf bm0 = *(const v16bf*)(wb + lane * 16);
        v16bf bm1 = *(const v16bf*)(wb + (32 + lane) * 16);

        int ww = w0 + m + kw - 1;
        union { f4 f[2]; v16bf v; } a;
        if (hok && (unsigned)ww < (unsigned)W_) {
          const __bf16* base =
              actb + ((size_t)dd * HW_ + (size_t)hh * W_ + ww) * 32 + gsel * 8;
          a.f[0] = *(const f4*)(base);        // K lo chunk  (8 halves)
          a.f[1] = *(const f4*)(base + 16);   // K hi chunk  (8 halves)
        } else {
          f4 z = {0.f, 0.f, 0.f, 0.f};
          a.f[0] = z; a.f[1] = z;
        }
        acc0 = __builtin_amdgcn_wmma_f32_16x16x32_bf16(
            false, a.v, false, bm0, (short)0, acc0, false, false);
        acc1 = __builtin_amdgcn_wmma_f32_16x16x32_bf16(
            false, a.v, false, bm1, (short)0, acc1, false, false);
      }
    }
  }

  // Stage 16x32 f32 tile to LDS (C/D layout: VGPR r -> M = r + 8*(lane>>4), N = lane&15)
  float* my = lds + wave * 512;
  int mrow = gsel * 8;
#pragma unroll
  for (int r = 0; r < 8; ++r) {
    my[(mrow + r) * 32 + (lane & 15)]      = acc0[r];
    my[(mrow + r) * 32 + (lane & 15) + 16] = acc1[r];
  }
  __syncthreads();

  // Coalesced epilogue: BN2 + residual + ReLU, w-contiguous stores.
  int mm = lane & 15;
  for (int it = 0; it < 16; ++it) {
    int n = it * 2 + gsel;
    float s2 = g2[n] * rsqrtf(v2[n] + EPS_);
    float bb = b2[n] - m2[n] * s2;
    float v = my[mm * 32 + n];
    size_t oidx = (((size_t)n * D_ + dz) * H_ + hy) * W_ + w0 + mm;
    float res = v * s2 + bb + xin[oidx];
    out[oidx] = fmaxf(res, 0.f);
  }
}

// ---------------------------------------------------------------------------
extern "C" void kernel_launch(void* const* d_in, const int* in_sizes, int n_in,
                              void* d_out, int out_size, void* d_ws, size_t ws_size,
                              hipStream_t stream) {
  const float* x  = (const float*)d_in[0];
  const float* g  = (const float*)d_in[1];
  const float* cw = (const float*)d_in[2];
  const float* g1 = (const float*)d_in[3];
  const float* b1 = (const float*)d_in[4];
  const float* m1 = (const float*)d_in[5];
  const float* v1 = (const float*)d_in[6];
  const float* g2 = (const float*)d_in[7];
  const float* b2 = (const float*)d_in[8];
  const float* m2 = (const float*)d_in[9];
  const float* v2 = (const float*)d_in[10];
  float* out = (float*)d_out;

  char* ws = (char*)d_ws;
  // Layout: [xt 81.8MB][sga 81.8MB][knorm 1.04MB][wfrag 55KB]; act reuses xt.
  float*  xt   = (float*)(ws);
  float*  sga  = (float*)(ws + 81788928);
  float*  kn   = (float*)(ws + 163577856);
  __bf16* wf   = (__bf16*)(ws + 164642816);
  __bf16* actb = (__bf16*)(ws);  // reuse xt region once scans are done

  knorm_kernel<<<(4 * HW_ + 255) / 256, 256, 0, stream>>>(g, kn);
  transpose_kernel<<<C_ * H_, 256, 0, stream>>>(x, xt);
  wpack_kernel<<<7, 256, 0, stream>>>(cw, wf);

  sga_scan_h<<<C_ * H_, 64, 0, stream>>>(xt, kn, sga, 0, 0, 0);  // L->R (store)
  sga_scan_h<<<C_ * H_, 64, 0, stream>>>(xt, kn, sga, 1, 1, 1);  // R->L (max)
  sga_scan_v<<<C_ * W_, 64, 0, stream>>>(xt, kn, sga, 2, 0, 1);  // T->B (max)
  sga_scan_v<<<C_ * W_, 64, 0, stream>>>(xt, kn, sga, 3, 1, 1);  // B->T (max)

  bn1_relu_pack<<<HW_, 64, 0, stream>>>(sga, g1, b1, m1, v1, actb);

  conv_bn2_res_kernel<<<(DHW_ / 16) / 8, 256, 0, stream>>>(
      actb, wf, x, g2, b2, m2, v2, out);
}
